// MatchNet_33285996544534
// MI455X (gfx1250) — compile-verified, matrix operand order checked by reading, not run on hardware
//
#include <hip/hip_runtime.h>
#include <hip/hip_bf16.h>

#define B_N    1024
#define H_HIST 16384
#define S_DIM  32
#define HID    64
#define G_N    10
#define K_N    4
#define A_N    8
#define GA     80
#define NROWS  (B_N + H_HIST)   // 17408

#define HT     64                       // history tile (2 WMMA k-steps)
#define TQ     64                       // query rows per block (4 waves x 16)
#define NSPLIT 4                        // split-K over history
#define HSPL   (H_HIST / NSPLIT)        // 4096
#define PSTR   (128 + 64 * GA)          // per-partial stride: m[64] l[64] acc[64][80]
#define LOG2E  1.4426950408889634f

typedef __attribute__((ext_vector_type(16))) _Float16 v16h;
typedef __attribute__((ext_vector_type(8)))  float    v8f;

union FragA { v16h h; unsigned int u[8]; };

__device__ __forceinline__ unsigned pk16(float a, float b) {
    auto p = __builtin_amdgcn_cvt_pkrtz(a, b);   // v_cvt_pk_rtz_f16_f32 (__fp16 x2)
    return __builtin_bit_cast(unsigned, p);
}

// ---------------------------------------------------------------------------
// Encoder: 3-layer MLP (S->H, H->H, H->H with ReLU), fp32 in LDS, f16 out.
// Query rows (< B_N) are pre-scaled by log2(e) so attention logits come out of
// the WMMA already in log2 domain.
// grid = (NROWS/64, K), 256 threads; thread = (row, 16-out chunk).
// ---------------------------------------------------------------------------
__global__ __launch_bounds__(256) void encode_kernel(
    const float* __restrict__ state, const float* __restrict__ hist,
    const float* __restrict__ W0, const float* __restrict__ b0,
    const float* __restrict__ W1, const float* __restrict__ b1,
    const float* __restrict__ W2, const float* __restrict__ b2,
    _Float16* __restrict__ enc)
{
    __shared__ float sW[HID * HID];
    __shared__ float sB[HID];
    __shared__ float xs[64 * 33];
    __shared__ float hA[64 * 65];
    __shared__ float hB[64 * 65];

    const int k    = blockIdx.y;
    const int t    = threadIdx.x;
    const int row0 = blockIdx.x * 64;
    const int row  = t >> 2;
    const int ob   = (t & 3) * 16;

    for (int idx = t; idx < 64 * S_DIM; idx += 256) {
        int r = idx >> 5, i = idx & 31;
        int gr = row0 + r;
        float v = (gr < B_N) ? state[gr * S_DIM + i] : hist[(gr - B_N) * S_DIM + i];
        xs[r * 33 + i] = v;
    }
    // ---- layer 0: 32 -> 64 ----
    for (int idx = t; idx < S_DIM * HID; idx += 256) sW[idx] = W0[k * S_DIM * HID + idx];
    if (t < HID) sB[t] = b0[k * HID + t];
    __syncthreads();
    {
        float acc[16];
        #pragma unroll
        for (int j = 0; j < 16; ++j) acc[j] = sB[ob + j];
        for (int i = 0; i < S_DIM; ++i) {
            float xv = xs[row * 33 + i];
            #pragma unroll
            for (int j = 0; j < 16; ++j) acc[j] += xv * sW[i * HID + ob + j];
        }
        #pragma unroll
        for (int j = 0; j < 16; ++j) hA[row * 65 + ob + j] = fmaxf(acc[j], 0.f);
    }
    __syncthreads();
    // ---- layer 1: 64 -> 64 ----
    for (int idx = t; idx < HID * HID; idx += 256) sW[idx] = W1[k * HID * HID + idx];
    if (t < HID) sB[t] = b1[k * HID + t];
    __syncthreads();
    {
        float acc[16];
        #pragma unroll
        for (int j = 0; j < 16; ++j) acc[j] = sB[ob + j];
        for (int i = 0; i < HID; ++i) {
            float xv = hA[row * 65 + i];
            #pragma unroll
            for (int j = 0; j < 16; ++j) acc[j] += xv * sW[i * HID + ob + j];
        }
        #pragma unroll
        for (int j = 0; j < 16; ++j) hB[row * 65 + ob + j] = fmaxf(acc[j], 0.f);
    }
    __syncthreads();
    // ---- layer 2: 64 -> 64, store f16 (queries scaled by log2e) ----
    for (int idx = t; idx < HID * HID; idx += 256) sW[idx] = W2[k * HID * HID + idx];
    if (t < HID) sB[t] = b2[k * HID + t];
    __syncthreads();
    {
        float acc[16];
        #pragma unroll
        for (int j = 0; j < 16; ++j) acc[j] = sB[ob + j];
        for (int i = 0; i < HID; ++i) {
            float xv = hB[row * 65 + i];
            #pragma unroll
            for (int j = 0; j < 16; ++j) acc[j] += xv * sW[i * HID + ob + j];
        }
        const float osc = ((row0 + row) < B_N) ? LOG2E : 1.0f;
        _Float16* dst = enc + ((size_t)k * NROWS + (row0 + row)) * HID + ob;
        #pragma unroll
        for (int j = 0; j < 16; ++j) dst[j] = (_Float16)(fmaxf(acc[j], 0.f) * osc);
    }
}

// Transpose actions [H,80] f32 -> [80,H] f16 (WMMA A-operand friendly)
__global__ __launch_bounds__(256) void prep_v_kernel(
    const float* __restrict__ jah, _Float16* __restrict__ vt)
{
    int idx = blockIdx.x * 256 + threadIdx.x;
    if (idx < H_HIST * GA) {
        int h = idx / GA, c = idx % GA;
        vt[(size_t)c * H_HIST + h] = (_Float16)jah[idx];
    }
}

// ---------------------------------------------------------------------------
// Fused flash attention, transposed formulation:
//   S^T = K * Q^T  (history rows in C-layout regs, query index in lanes)
//   P^T packed in-register (cvt_pk_rtz) into the WMMA B-operand slots
//   acc^T = V^T * P^T
// Softmax state (m, l) is one scalar per lane; only cross-lane op is a single
// xor-16 half exchange per reduction. Zero LDS.
// grid = (B/64, K, NSPLIT), 128 threads (4 waves x 16 query rows).
// ---------------------------------------------------------------------------
__global__ __launch_bounds__(128) void flash_attn_kernel(
    const _Float16* __restrict__ enc, const _Float16* __restrict__ vt,
    float* __restrict__ part)
{
    const int btile = blockIdx.x;
    const int k     = blockIdx.y;
    const int split = blockIdx.z;
    const int wave  = threadIdx.x >> 5;
    const int lane  = threadIdx.x & 31;
    const int hf    = lane >> 4;           // lane half
    const int n     = lane & 15;
    const int kb    = hf * 8;

    int kp[8];
    #pragma unroll
    for (int j = 0; j < 8; ++j) kp[j] = kb + (j & 3) * 2 + (j >> 2) * 16;

    const _Float16* enc_k = enc + (size_t)k * NROWS * HID;
    const _Float16* qrow  = enc_k + (size_t)(btile * TQ + wave * 16 + n) * HID;
    const _Float16* kh    = enc_k + (size_t)B_N * HID;   // enc_h rows

    FragA qa0, qa1;                        // Q^T B-fragments, d=[0,32) / [32,64)
    #pragma unroll
    for (int j = 0; j < 8; ++j) {
        qa0.u[j] = *(const unsigned int*)(qrow + kp[j]);
        qa1.u[j] = *(const unsigned int*)(qrow + 32 + kp[j]);
    }

    const v8f vzero = {0.f, 0.f, 0.f, 0.f, 0.f, 0.f, 0.f, 0.f};
    v8f acc[5];                            // acc^T tiles: regs=action col, lanes=query
    #pragma unroll
    for (int c = 0; c < 5; ++c) acc[c] = vzero;
    float mval = -3.0e38f;                 // log2-domain running max (per query row)
    float lval = 0.f;

    const int hbeg = split * HSPL;
    const int hend = hbeg + HSPL;

    for (int h0 = hbeg; h0 < hend; h0 += HT) {
        // prefetch next K tile rows (128B rows -> 2 prefetches cover 64 rows)
        __builtin_prefetch(kh + (size_t)(h0 + HT + lane * 2) * HID, 0, 1);
        __builtin_prefetch(kh + (size_t)(h0 + HT + lane * 2 + 1) * HID, 0, 1);

        // ---- S^T = K * Q^T : four 16-row history subtiles ----
        v8f s[4];
        #pragma unroll
        for (int t = 0; t < 4; ++t) {
            FragA kf0, kf1;
            const _Float16* kr = kh + (size_t)(h0 + t * 16 + n) * HID;
            #pragma unroll
            for (int j = 0; j < 8; ++j) {
                kf0.u[j] = *(const unsigned int*)(kr + kp[j]);
                kf1.u[j] = *(const unsigned int*)(kr + 32 + kp[j]);
            }
            s[t] = __builtin_amdgcn_wmma_f32_16x16x32_f16(false, kf0.h, false, qa0.h, (short)0, vzero, false, false);
            s[t] = __builtin_amdgcn_wmma_f32_16x16x32_f16(false, kf1.h, false, qa1.h, (short)0, s[t],  false, false);
        }

        // ---- online softmax (log2 domain); each lane owns one query row,
        //      holding 32 of the 64 logits; partner lane (xor 16) has the rest.
        float tt[8];
        #pragma unroll
        for (int r = 0; r < 8; ++r)
            tt[r] = fmaxf(fmaxf(s[0][r], s[1][r]), fmaxf(s[2][r], s[3][r]));
        float a = fmaxf(fmaxf(fmaxf(tt[0], tt[1]), fmaxf(tt[2], tt[3])),
                        fmaxf(fmaxf(tt[4], tt[5]), fmaxf(tt[6], tt[7])));
        a = fmaxf(a, __shfl_xor(a, 16));
        const float nm = fmaxf(mval, a);
        const float sc = __builtin_amdgcn_exp2f(mval - nm);

        float e0[8], e1[8], e2[8], e3[8], ss[8];
        #pragma unroll
        for (int r = 0; r < 8; ++r) {
            e0[r] = __builtin_amdgcn_exp2f(s[0][r] - nm);
            e1[r] = __builtin_amdgcn_exp2f(s[1][r] - nm);
            e2[r] = __builtin_amdgcn_exp2f(s[2][r] - nm);
            e3[r] = __builtin_amdgcn_exp2f(s[3][r] - nm);
            ss[r] = (e0[r] + e1[r]) + (e2[r] + e3[r]);
        }
        float rs = ((ss[0] + ss[1]) + (ss[2] + ss[3])) + ((ss[4] + ss[5]) + (ss[6] + ss[7]));
        rs += __shfl_xor(rs, 16);
        lval = lval * sc + rs;
        mval = nm;
        #pragma unroll
        for (int c = 0; c < 5; ++c) acc[c] *= sc;

        // ---- pack P^T B-fragments fully in-register (no LDS, no shuffles) ----
        FragA bP0, bP1;
        #pragma unroll
        for (int j2 = 0; j2 < 4; ++j2) {
            bP0.u[j2]     = pk16(e0[2 * j2], e0[2 * j2 + 1]);
            bP0.u[4 + j2] = pk16(e1[2 * j2], e1[2 * j2 + 1]);
            bP1.u[j2]     = pk16(e2[2 * j2], e2[2 * j2 + 1]);
            bP1.u[4 + j2] = pk16(e3[2 * j2], e3[2 * j2 + 1]);
        }

        // ---- acc^T += V^T * P^T over the 80 action columns (5 tiles x k=64) ----
        #pragma unroll
        for (int c = 0; c < 5; ++c) {
            FragA va0, va1;
            const _Float16* vr = vt + (size_t)(c * 16 + n) * H_HIST + h0;
            #pragma unroll
            for (int j = 0; j < 8; ++j) {
                va0.u[j] = *(const unsigned int*)(vr + kp[j]);
                va1.u[j] = *(const unsigned int*)(vr + 32 + kp[j]);
            }
            acc[c] = __builtin_amdgcn_wmma_f32_16x16x32_f16(false, va0.h, false, bP0.h, (short)0, acc[c], false, false);
            acc[c] = __builtin_amdgcn_wmma_f32_16x16x32_f16(false, va1.h, false, bP1.h, (short)0, acc[c], false, false);
        }
    }

    // ---- write split partials: m[64] (log2 domain), l[64], acc[64][80] ----
    // acc^T tile c: reg r -> action col c*16 + kb + r, lane -> query row n.
    float* pr = part + (((size_t)k * 16 + btile) * NSPLIT + split) * PSTR;
    const int row = wave * 16 + n;
    float* pa = pr + 128 + (size_t)row * GA;
    #pragma unroll
    for (int c = 0; c < 5; ++c)
        *(v8f*)(pa + c * 16 + kb) = acc[c];
    if (hf == 0) {
        pr[row]      = mval;
        pr[64 + row] = lval;
    }
}

// ---------------------------------------------------------------------------
// Combine: merge split softmax partials (log2 domain), apply softmax(assign).
// grid = B, 96 threads (80 active = G*A).
// ---------------------------------------------------------------------------
__global__ __launch_bounds__(96) void combine_kernel(
    const float* __restrict__ part, const float* __restrict__ assign,
    float* __restrict__ out)
{
    const int b = blockIdx.x;
    const int t = threadIdx.x;
    if (t >= GA) return;
    const int g     = t / A_N;
    const int btile = b >> 6;
    const int r     = b & 63;

    float aw[K_N];
    float am = -3.0e38f;
    #pragma unroll
    for (int k = 0; k < K_N; ++k) { aw[k] = assign[g * K_N + k]; am = fmaxf(am, aw[k]); }
    float asum = 0.f;
    #pragma unroll
    for (int k = 0; k < K_N; ++k) { aw[k] = __expf(aw[k] - am); asum += aw[k]; }

    float o = 0.f;
    #pragma unroll
    for (int k = 0; k < K_N; ++k) {
        const float* pk = part + (((size_t)k * 16 + btile) * NSPLIT) * PSTR;
        float mm = -3.0e38f;
        #pragma unroll
        for (int s = 0; s < NSPLIT; ++s) mm = fmaxf(mm, pk[s * PSTR + r]);
        float lsum = 0.f, vsum = 0.f;
        #pragma unroll
        for (int s = 0; s < NSPLIT; ++s) {
            float w = __builtin_amdgcn_exp2f(pk[s * PSTR + r] - mm);  // log2 domain
            lsum += w * pk[s * PSTR + 64 + r];
            vsum += w * pk[s * PSTR + 128 + r * GA + t];
        }
        o += (aw[k] / asum) * (vsum / lsum);
    }
    out[(size_t)b * GA + t] = o;
}

extern "C" void kernel_launch(void* const* d_in, const int* in_sizes, int n_in,
                              void* d_out, int out_size, void* d_ws, size_t ws_size,
                              hipStream_t stream) {
    const float* state  = (const float*)d_in[0];
    const float* hist   = (const float*)d_in[1];
    const float* jah    = (const float*)d_in[2];
    const float* W0     = (const float*)d_in[3];
    const float* b0     = (const float*)d_in[4];
    const float* W1     = (const float*)d_in[5];
    const float* b1     = (const float*)d_in[6];
    const float* W2     = (const float*)d_in[7];
    const float* b2     = (const float*)d_in[8];
    const float* assign = (const float*)d_in[9];
    float* out = (float*)d_out;

    char* ws = (char*)d_ws;
    const size_t enc_bytes = (size_t)K_N * NROWS * HID * sizeof(_Float16);   // ~8.9 MB
    const size_t vt_bytes  = (size_t)GA * H_HIST * sizeof(_Float16);         // ~2.6 MB
    _Float16* enc = (_Float16*)ws;
    _Float16* vtp = (_Float16*)(ws + enc_bytes);
    float*    part = (float*)(ws + enc_bytes + vt_bytes);                    // ~5.4 MB

    encode_kernel<<<dim3(NROWS / 64, K_N), 256, 0, stream>>>(
        state, hist, W0, b0, W1, b1, W2, b2, enc);
    prep_v_kernel<<<(H_HIST * GA + 255) / 256, 256, 0, stream>>>(jah, vtp);
    flash_attn_kernel<<<dim3(B_N / TQ, K_N, NSPLIT), 128, 0, stream>>>(enc, vtp, part);
    combine_kernel<<<B_N, 96, 0, stream>>>(part, assign, out);
}
